// BiLSTM_CRF_38053410243254
// MI455X (gfx1250) — matrix operand order repaired and checked by reading, not compiled, not run on hardware
//
#include <hip/hip_runtime.h>
#include <hip/hip_bf16.h>
#include <math.h>

// ---------------------------------------------------------------------------
// BiLSTM-CRF forward loss for MI455X (gfx1250), wave32 + WMMA bf16.
// B=64, T=256, EMB=300(pad 320), HW=128, HP=8, K=6 tags.
// ---------------------------------------------------------------------------

typedef __attribute__((ext_vector_type(16))) __bf16 v16bf;
typedef __attribute__((ext_vector_type(8)))  __bf16 v8bf;
typedef __attribute__((ext_vector_type(8)))  float  v8f;

#define BT        16384   // B*T rows
#define BATCH     64
#define SEQ       256
#define KTAGS     6
#define START_TAG 3
#define END_TAG   4
#define PAD_TAG   5

union V16U { v16bf v; v8bf h[2]; };

__device__ __forceinline__ float sigf(float x) { return 1.0f / (1.0f + __expf(-x)); }

// A-matrix 16x32 bf16 fragment (ISA 7.12.2): lane<16 -> K 0..7 & 16..23,
// lane>=16 -> K 8..15 & 24..31 for row M = mbase + (lane&15).
__device__ __forceinline__ v16bf load_a16x32(const __bf16* base, int ld,
                                             int mbase, int k0, int lane) {
  int r    = mbase + (lane & 15);
  int koff = (lane & 16) ? 8 : 0;
  const __bf16* p = base + (size_t)r * ld + k0 + koff;
  V16U u;
  u.h[0] = *(const v8bf*)(p);
  u.h[1] = *(const v8bf*)(p + 16);
  return u.v;
}

// B-matrix 32x16 bf16 fragment: lane<16 holds K 0..7 & 16..23, lane>=16 holds K 8..15 & 24..31
// for column N = nbase + (lane&15). W stored [N,K] row-major -> contiguous.
__device__ __forceinline__ v16bf load_b32x16(const __bf16* W, int ld,
                                             int nbase, int k0, int lane) {
  int nr   = nbase + (lane & 15);
  int koff = (lane & 16) ? 8 : 0;
  const __bf16* p = W + (size_t)nr * ld + k0 + koff;
  V16U u;
  u.h[0] = *(const v8bf*)(p);
  u.h[1] = *(const v8bf*)(p + 16);
  return u.v;
}

// --------------------------- gather / convert ------------------------------

__global__ __launch_bounds__(128) void k_gather(const int* __restrict__ words,
                                                const int* __restrict__ pos,
                                                const float* __restrict__ emb_w,
                                                __bf16* __restrict__ weA,   // [BT,320]
                                                __bf16* __restrict__ peA) { // [BT,64]
  int row = blockIdx.x;
  int wid = words[row];
  int pid = pos[row];
  for (int k = threadIdx.x; k < 320; k += 128) {
    float v = (k < 300) ? emb_w[(size_t)wid * 300 + k] : 0.0f;
    weA[(size_t)row * 320 + k] = (__bf16)v;
  }
  for (int k = threadIdx.x; k < 64; k += 128)
    peA[(size_t)row * 64 + k] = (__bf16)((k == pid) ? 1.0f : 0.0f);
}

__global__ __launch_bounds__(256) void k_cvt_w(const float* __restrict__ src,
                                               __bf16* __restrict__ dst,
                                               int N, int K, int Kp) {
  int idx = blockIdx.x * 256 + threadIdx.x;
  if (idx >= N * Kp) return;
  int n = idx / Kp, k = idx - n * Kp;
  dst[idx] = (__bf16)((k < K) ? src[(size_t)n * K + k] : 0.0f);
}

// --------------------------- generic WMMA GEMM -----------------------------
// C[M,NT*16*gridDim.y] = A[M,KP](bf16) x W[N,KP](bf16)^T + bias, opt. ReLU.
// Compile-time NT (N-tiles per wave) and KP so all accumulator indexing is
// static: no v_movrel, WMMAs issue back-to-back from fixed VGPR octets.
// 8 waves/block; wave computes a 16 x (NT*16) slab.

template <int NT, int KP>
__global__ __launch_bounds__(256) void k_gemm_bf16(
    const __bf16* __restrict__ A, const __bf16* __restrict__ W,
    const float* __restrict__ bias, float* __restrict__ C,
    int lda, int ldw, int ldc, int do_relu) {
  int lane  = threadIdx.x & 31;
  int wave  = threadIdx.x >> 5;
  int mbase = (blockIdx.x * 8 + wave) * 16;
  int nt0   = blockIdx.y * NT;

  v8f acc[NT];
#pragma unroll
  for (int j = 0; j < NT; ++j)
#pragma unroll
    for (int r = 0; r < 8; ++r) acc[j][r] = 0.0f;

#pragma unroll
  for (int k0 = 0; k0 < KP; k0 += 32) {
    v16bf a = load_a16x32(A, lda, mbase, k0, lane);
    __builtin_prefetch(A + (size_t)(mbase + (lane & 15)) * lda + k0 + 128, 0, 1);
#pragma unroll
    for (int j = 0; j < NT; ++j) {
      v16bf b = load_b32x16(W, ldw, (nt0 + j) * 16, k0, lane);
      acc[j] = __builtin_amdgcn_wmma_f32_16x16x32_bf16(
          false, a, false, b, (short)0, acc[j], false, false);
    }
  }

  int hi8 = (lane & 16) ? 8 : 0;
  int cl  = lane & 15;
#pragma unroll
  for (int j = 0; j < NT; ++j) {
    int col  = (nt0 + j) * 16 + cl;
    float bv = bias[col];
#pragma unroll
    for (int r = 0; r < 8; ++r) {
      float v = acc[j][r] + bv;
      if (do_relu) v = fmaxf(v, 0.0f);
      C[(size_t)(mbase + r + hi8) * ldc + col] = v;
    }
  }
}

// --------------------------- word BiLSTM recurrence ------------------------
// blockIdx.x: 0=fwd, 1=bwd. 32 waves; wave owns (batch-tile, hidden-tile) and
// computes i/f/g/o tiles for the same hidden cols -> c stays in VGPRs.
// Wh (128 KB) is staged into LDS once; per-step B fragments are ds_load_b128
// with a 2-deep software pipeline so each WMMA overlaps the next fragment's
// load. A laundered zero offset keeps the ds addresses opaque to LICM (no
// 128-VGPR hoist + scratch spill) while staying in the LDS address space.

#define LSTM_LDS_BYTES ((64 * 128 + 512 * 128) * 2)   // hbuf + whs = 147456 B

__global__ __launch_bounds__(1024) void k_word_lstm(
    const float* __restrict__ xg_f, const float* __restrict__ xg_b,   // [BT,512]
    const __bf16* __restrict__ Wh_f, const __bf16* __restrict__ Wh_b, // [512,128]
    __bf16* __restrict__ wo) {                                        // [BT,256]
  const int rev = blockIdx.x;
  const float*  xg = rev ? xg_b : xg_f;
  const __bf16* Wh = rev ? Wh_b : Wh_f;
  const int coff   = rev ? 128 : 0;

  extern __shared__ __bf16 smem[];
  __bf16* hbuf = smem;              // [64][128]  previous h (16 KB)
  __bf16* whs  = smem + 64 * 128;   // [512][128] staged Wh  (128 KB)

  int tid   = threadIdx.x;
  int lane  = tid & 31;
  int wave  = tid >> 5;
  int mbase = (wave >> 3) * 16;      // batch tile
  int nbase = (wave & 7) * 16;       // hidden tile
  int hi8   = (lane & 16) ? 8 : 0;
  int cl    = lane & 15;

  for (int i = tid; i < 64 * 128; i += 1024) hbuf[i] = (__bf16)0.0f;
  for (int i = tid; i < (512 * 128) / 8; i += 1024)   // 16B chunks
    ((v8bf*)whs)[i] = ((const v8bf*)Wh)[i];
  float c[8];
#pragma unroll
  for (int r = 0; r < 8; ++r) c[r] = 0.0f;
  __syncthreads();

  for (int t = 0; t < SEQ; ++t) {
    int tt = rev ? (SEQ - 1 - t) : t;

    // gate pre-activations initialized with the precomputed input projection
    v8f acc[4];  // i, f, g, o for this (batch-tile, hidden-tile)
#pragma unroll
    for (int g = 0; g < 4; ++g)
#pragma unroll
      for (int r = 0; r < 8; ++r)
        acc[g][r] = xg[((size_t)(mbase + r + hi8) * SEQ + tt) * 512 +
                       g * 128 + nbase + cl];

    // all 4 A-fragments (h from LDS) up front
    v16bf afr[4];
#pragma unroll
    for (int k = 0; k < 4; ++k)
      afr[k] = load_a16x32(hbuf, 128, mbase, k * 32, lane);

    // opaque zero offset: keeps B loads un-hoistable but still in LDS space
    int zoff = 0;
    asm volatile("" : "+v"(zoff));

    // 16 (gate,k) WMMAs with 2-deep B-fragment pipeline from LDS
    v16bf bcur = load_b32x16(whs, 128, zoff + nbase, 0, lane);
#pragma unroll
    for (int idx = 0; idx < 16; ++idx) {
      const int g = idx >> 2, k = idx & 3;
      v16bf bnext;
      if (idx < 15) {
        const int g2 = (idx + 1) >> 2, k2 = (idx + 1) & 3;
        bnext = load_b32x16(whs, 128, zoff + g2 * 128 + nbase, k2 * 32, lane);
      }
      acc[g] = __builtin_amdgcn_wmma_f32_16x16x32_bf16(
          false, afr[k], false, bcur, (short)0, acc[g], false, false);
      if (idx < 15) bcur = bnext;
    }
    __syncthreads();   // all waves done reading hbuf(t-1)

#pragma unroll
    for (int r = 0; r < 8; ++r) {
      float iv = sigf(acc[0][r]);
      float fv = sigf(acc[1][r]);
      float gv = tanhf(acc[2][r]);
      float ov = sigf(acc[3][r]);
      c[r] = fv * c[r] + iv * gv;
      float hv = ov * tanhf(c[r]);
      int b = mbase + r + hi8;
      hbuf[b * 128 + nbase + cl] = (__bf16)hv;
      wo[((size_t)b * SEQ + tt) * 256 + coff + nbase + cl] = (__bf16)hv;
    }
    __syncthreads();   // hbuf(t) visible before next step's reads
  }
}

// --------------------------- POS BiLSTM (H=8, scalar) ----------------------

__global__ __launch_bounds__(64) void k_pos_lstm(
    const float* __restrict__ xg_f, const float* __restrict__ xg_b, // [BT,32]
    const float* __restrict__ Whf, const float* __restrict__ Whb,   // [32,8]
    float* __restrict__ po) {                                       // [BT,16]
  int rev = blockIdx.x;
  const float* xg = rev ? xg_b : xg_f;
  const float* Wh = rev ? Whb : Whf;
  int coff = rev ? 8 : 0;
  int b = threadIdx.x;

  __shared__ float wsh[256];
  for (int i = threadIdx.x; i < 256; i += 64) wsh[i] = Wh[i];
  __syncthreads();

  float h[8], c[8];
#pragma unroll
  for (int k = 0; k < 8; ++k) { h[k] = 0.0f; c[k] = 0.0f; }
  for (int t = 0; t < SEQ; ++t) {
    int tt = rev ? (SEQ - 1 - t) : t;
    const float* xr = xg + (size_t)(b * SEQ + tt) * 32;
    float g[32];
#pragma unroll
    for (int j = 0; j < 32; ++j) {
      float s = xr[j];
#pragma unroll
      for (int k = 0; k < 8; ++k) s += wsh[j * 8 + k] * h[k];
      g[j] = s;
    }
#pragma unroll
    for (int k = 0; k < 8; ++k) {
      float iv = sigf(g[k]), fv = sigf(g[8 + k]);
      float gv = tanhf(g[16 + k]), ov = sigf(g[24 + k]);
      c[k] = fv * c[k] + iv * gv;
      h[k] = ov * tanhf(c[k]);
      po[(size_t)(b * SEQ + tt) * 16 + coff + k] = h[k];
    }
  }
}

// --------------------------- small projections -----------------------------

__global__ __launch_bounds__(256) void k_ph(const float* __restrict__ po,
                                            const float* __restrict__ phW,
                                            const float* __restrict__ phb,
                                            float* __restrict__ ph) {
  int idx = blockIdx.x * 256 + threadIdx.x;   // BT*8
  if (idx >= BT * 8) return;
  int row = idx >> 3, j = idx & 7;
  float s = phb[j];
#pragma unroll
  for (int k = 0; k < 16; ++k) s += po[(size_t)row * 16 + k] * phW[j * 16 + k];
  ph[idx] = fmaxf(s, 0.0f);
}

__global__ __launch_bounds__(256) void k_feats(const float* __restrict__ wh,
                                               const float* __restrict__ ph,
                                               const float* __restrict__ tagW,
                                               const float* __restrict__ tagb,
                                               float* __restrict__ feats) {
  int idx = blockIdx.x * 256 + threadIdx.x;   // BT*6
  if (idx >= BT * KTAGS) return;
  int row = idx / KTAGS, j = idx - row * KTAGS;
  float s = tagb[j];
  for (int k = 0; k < 128; ++k) s += wh[(size_t)row * 128 + k] * tagW[j * 136 + k];
#pragma unroll
  for (int k = 0; k < 8; ++k)   s += ph[(size_t)row * 8 + k]  * tagW[j * 136 + 128 + k];
  feats[idx] = fmaxf(s, 0.0f);
}

// --------------------------- CRF NLL (1 block, lane = batch) ---------------

__global__ __launch_bounds__(64) void k_crf(const float* __restrict__ feats,
                                            const int* __restrict__ words,
                                            const int* __restrict__ labels,
                                            const float* __restrict__ trans,
                                            float* __restrict__ out) {
  __shared__ float tn[KTAGS][KTAGS];
  __shared__ float red[BATCH];
  int tid = threadIdx.x;

  if (tid < KTAGS) {            // softmax over next-tag axis (axis 0)
    int col = tid;
    float mx = -1e30f;
    for (int r = 0; r < KTAGS; ++r) mx = fmaxf(mx, trans[r * KTAGS + col]);
    float s = 0.0f;
    for (int r = 0; r < KTAGS; ++r) s += __expf(trans[r * KTAGS + col] - mx);
    for (int r = 0; r < KTAGS; ++r)
      tn[r][col] = (col == END_TAG) ? -100.0f
                                    : (__expf(trans[r * KTAGS + col] - mx) / s);
  }
  __syncthreads();

  int b = tid;
  int len = 0;
  for (int t = 0; t < SEQ; ++t) len += (words[b * SEQ + t] != 0) ? 1 : 0;

  float alpha[KTAGS];
#pragma unroll
  for (int j = 0; j < KTAGS; ++j) alpha[j] = (j == START_TAG) ? 0.0f : -100.0f;
  for (int t = 0; t < SEQ; ++t) {
    if (t < len) {
      float na[KTAGS];
#pragma unroll
      for (int nx = 0; nx < KTAGS; ++nx) {
        float mx = -1e30f;
#pragma unroll
        for (int pv = 0; pv < KTAGS; ++pv) mx = fmaxf(mx, alpha[pv] + tn[nx][pv]);
        float s = 0.0f;
#pragma unroll
        for (int pv = 0; pv < KTAGS; ++pv) s += __expf(alpha[pv] + tn[nx][pv] - mx);
        na[nx] = mx + __logf(s) + feats[(size_t)(b * SEQ + t) * KTAGS + nx];
      }
#pragma unroll
      for (int j = 0; j < KTAGS; ++j) alpha[j] = na[j];
    }
  }
  float mx = -1e30f;
#pragma unroll
  for (int j = 0; j < KTAGS; ++j) mx = fmaxf(mx, alpha[j] + trans[END_TAG * KTAGS + j]);
  float s = 0.0f;
#pragma unroll
  for (int j = 0; j < KTAGS; ++j) s += __expf(alpha[j] + trans[END_TAG * KTAGS + j] - mx);
  float fwd = mx + __logf(s);

  float gold = 0.0f, valid = 1.0f;
  int prev = START_TAG;
  for (int t = 0; t < SEQ; ++t) {
    int lab = labels[b * SEQ + t];
    valid *= (lab != PAD_TAG) ? 1.0f : 0.0f;
    gold += valid * (feats[(size_t)(b * SEQ + t) * KTAGS + lab] + tn[lab][prev]);
    prev = lab;
  }
  gold += trans[END_TAG * KTAGS + labels[b * SEQ + SEQ - 1]];

  red[b] = fwd - gold;
  __syncthreads();
  if (tid == 0) {
    float acc = 0.0f;
    for (int i = 0; i < BATCH; ++i) acc += red[i];
    out[0] = acc / (float)BATCH;
  }
}

// --------------------------- host orchestration ----------------------------

extern "C" void kernel_launch(void* const* d_in, const int* in_sizes, int n_in,
                              void* d_out, int out_size, void* d_ws, size_t ws_size,
                              hipStream_t stream) {
  (void)in_sizes; (void)n_in; (void)out_size; (void)ws_size;
  const int*   words = (const int*)d_in[0];
  const int*   pos   = (const int*)d_in[1];
  const int*   labs  = (const int*)d_in[2];
  const float* emb_w = (const float*)d_in[3];
  const float* wlfWi = (const float*)d_in[5];
  const float* wlfWh = (const float*)d_in[6];
  const float* wlfb  = (const float*)d_in[7];
  const float* wlbWi = (const float*)d_in[8];
  const float* wlbWh = (const float*)d_in[9];
  const float* wlbb  = (const float*)d_in[10];
  const float* plfWi = (const float*)d_in[11];
  const float* plfWh = (const float*)d_in[12];
  const float* plfb  = (const float*)d_in[13];
  const float* plbWi = (const float*)d_in[14];
  const float* plbWh = (const float*)d_in[15];
  const float* plbb  = (const float*)d_in[16];
  const float* whW   = (const float*)d_in[17];
  const float* whb   = (const float*)d_in[18];
  const float* phW   = (const float*)d_in[19];
  const float* phb   = (const float*)d_in[20];
  const float* tagW  = (const float*)d_in[21];
  const float* tagb  = (const float*)d_in[22];
  const float* trans = (const float*)d_in[23];

  char* w = (char*)d_ws;
  size_t off = 0;
  auto carve = [&](size_t bytes) -> void* {
    off = (off + 255) & ~(size_t)255;
    void* p = w + off;
    off += bytes;
    return p;
  };
  __bf16* weA    = (__bf16*)carve((size_t)BT * 320 * 2);
  __bf16* peA    = (__bf16*)carve((size_t)BT * 64 * 2);
  __bf16* wWi_f  = (__bf16*)carve((size_t)512 * 320 * 2);
  __bf16* wWi_b  = (__bf16*)carve((size_t)512 * 320 * 2);
  __bf16* wWh_f  = (__bf16*)carve((size_t)512 * 128 * 2);
  __bf16* wWh_b  = (__bf16*)carve((size_t)512 * 128 * 2);
  __bf16* pWi_f  = (__bf16*)carve((size_t)32 * 64 * 2);
  __bf16* pWi_b  = (__bf16*)carve((size_t)32 * 64 * 2);
  __bf16* whWb   = (__bf16*)carve((size_t)128 * 256 * 2);
  float*  xg_wf  = (float*)carve((size_t)BT * 512 * 4);
  float*  xg_wb  = (float*)carve((size_t)BT * 512 * 4);
  float*  xg_pf  = (float*)carve((size_t)BT * 32 * 4);
  float*  xg_pb  = (float*)carve((size_t)BT * 32 * 4);
  __bf16* wo     = (__bf16*)carve((size_t)BT * 256 * 2);
  float*  whO    = (float*)carve((size_t)BT * 128 * 4);
  float*  po     = (float*)carve((size_t)BT * 16 * 4);
  float*  phO    = (float*)carve((size_t)BT * 8 * 4);
  float*  feats  = (float*)carve((size_t)BT * 6 * 4);

  // 1) gathers + weight conversion (f32 -> padded bf16)
  k_gather<<<BT, 128, 0, stream>>>(words, pos, emb_w, weA, peA);
  auto cvt = [&](const float* s, __bf16* d, int N, int K, int Kp) {
    int n = N * Kp;
    k_cvt_w<<<(n + 255) / 256, 256, 0, stream>>>(s, d, N, K, Kp);
  };
  cvt(wlfWi, wWi_f, 512, 300, 320);
  cvt(wlbWi, wWi_b, 512, 300, 320);
  cvt(wlfWh, wWh_f, 512, 128, 128);
  cvt(wlbWh, wWh_b, 512, 128, 128);
  cvt(plfWi, pWi_f, 32, 46, 64);
  cvt(plbWi, pWi_b, 32, 46, 64);
  cvt(whW,   whWb, 128, 256, 256);

  // 2) input-projection GEMMs (WMMA): M=16384 -> grid.x = 16384/16/8 = 128
  k_gemm_bf16<4, 320><<<dim3(128, 8), 256, 0, stream>>>(
      weA, wWi_f, wlfb, xg_wf, 320, 320, 512, 0);        // N=512
  k_gemm_bf16<4, 320><<<dim3(128, 8), 256, 0, stream>>>(
      weA, wWi_b, wlbb, xg_wb, 320, 320, 512, 0);        // N=512
  k_gemm_bf16<2, 64><<<dim3(128, 1), 256, 0, stream>>>(
      peA, pWi_f, plfb, xg_pf, 64, 64, 32, 0);           // N=32
  k_gemm_bf16<2, 64><<<dim3(128, 1), 256, 0, stream>>>(
      peA, pWi_b, plbb, xg_pb, 64, 64, 32, 0);           // N=32

  // 3) recurrences (dynamic LDS: 16 KB hbuf + 128 KB staged Wh)
  k_word_lstm<<<2, 1024, LSTM_LDS_BYTES, stream>>>(xg_wf, xg_wb, wWh_f, wWh_b, wo);
  k_pos_lstm<<<2, 64, 0, stream>>>(xg_pf, xg_pb, plfWh, plbWh, po);

  // 4) hidden projections
  k_gemm_bf16<4, 256><<<dim3(128, 2), 256, 0, stream>>>(
      wo, whWb, whb, whO, 256, 256, 128, 1);             // N=128, ReLU
  k_ph<<<(BT * 8 + 255) / 256, 256, 0, stream>>>(po, phW, phb, phO);
  k_feats<<<(BT * 6 + 255) / 256, 256, 0, stream>>>(whO, phO, tagW, tagb, feats);

  // 5) CRF negative log-likelihood -> scalar
  k_crf<<<1, 64, 0, stream>>>(feats, words, labs, trans, (float*)d_out);
}
